// VQVAE_61383672594730
// MI455X (gfx1250) — matrix operand order repaired and checked by reading, not compiled, not run on hardware
//
#include <hip/hip_runtime.h>
#include <hip/hip_bf16.h>
#include <math.h>

typedef __bf16 bf16_t;
typedef __attribute__((ext_vector_type(16))) __bf16 v16bf;
typedef __attribute__((ext_vector_type(8)))  __bf16 v8bf;
typedef __attribute__((ext_vector_type(8)))  float  v8f;

#define B_   128
#define T_   64
#define V_   10000
#define NI_  512
#define EH_  640
#define DH_  1024
#define NZ_  128
#define ND_  8
#define KC_  1024
#define DC_  160

#define VPAD_ 10112   // V_ rounded up to block-tile N (128)

// ---------------------------------------------------------------------------
// bf16 WMMA GEMM:  Cout[M,N] = A[M,K] * W[N,K]^T (+bias[n]) (+Cin[m,n])
// Block = 256 threads = 8 waves arranged 2(M) x 4(N); each wave owns a 32x32
// tile computed as 2x2 WMMA 16x16x32 sub-tiles (A frags reused across N,
// B frags reused across M).  Block tile = 64(M) x 128(N).
// Requirements: K % 32 == 0; M % 64 == 0; W has >= ceil(N/128)*128 rows
// (zero-padded); stores are guarded by n < N.
// ---------------------------------------------------------------------------
__device__ __forceinline__ v16bf load_afrag(const bf16_t* __restrict__ row,
                                            int k0, int hi) {
  // A frag, contiguous k order used here (same as B): lane-lo K=k0..k0+15,
  //                        lane-hi K=k0+16..k0+31
  v8bf x0 = *(const v8bf*)(row + k0 + hi * 16);
  v8bf x1 = *(const v8bf*)(row + k0 + hi * 16 + 8);
  v16bf r;
#pragma unroll
  for (int i = 0; i < 8; ++i) { r[i] = x0[i]; r[8 + i] = x1[i]; }
  return r;
}

__global__ __launch_bounds__(256) void k_gemm_bf16(
    const bf16_t* __restrict__ A, long lda,
    const bf16_t* __restrict__ W,          // (Npad,K), row stride K
    const float*  __restrict__ bias,       // may be null
    const float*  __restrict__ Cin, long ldcin,  // may be null
    float* __restrict__ Cout, long ldc,
    int M, int N, int K)
{
  const int wave = threadIdx.x >> 5;
  const int lane = threadIdx.x & 31;
  const int hi   = lane >> 4;
  const int l15  = lane & 15;

  const int mBase = blockIdx.y * 64  + (wave & 1) * 32;
  const int nBase = blockIdx.x * 128 + (wave >> 1) * 32;

  const bf16_t* __restrict__ Arow0 = A + (size_t)(mBase + l15)      * lda;
  const bf16_t* __restrict__ Arow1 = A + (size_t)(mBase + 16 + l15) * lda;
  const bf16_t* __restrict__ Wrow0 = W + (size_t)(nBase + l15)      * (size_t)K;
  const bf16_t* __restrict__ Wrow1 = W + (size_t)(nBase + 16 + l15) * (size_t)K;

  v8f acc00 = {}, acc01 = {}, acc10 = {}, acc11 = {};
  for (int k0 = 0; k0 < K; k0 += 32) {
    __builtin_prefetch(Wrow0 + k0 + 128, 0, 1);   // global_prefetch_b8
    __builtin_prefetch(Wrow1 + k0 + 128, 0, 1);
    v16bf a0 = load_afrag(Arow0, k0, hi);
    v16bf a1 = load_afrag(Arow1, k0, hi);
    // B frag, same contiguous k order as A: lane-lo K=k0..k0+15, lane-hi K=k0+16..k0+31,
    // column N = lane&15 -> one contiguous 32B load from W row n.
    v16bf b0 = *(const v16bf*)(Wrow0 + k0 + hi * 16);
    v16bf b1 = *(const v16bf*)(Wrow1 + k0 + hi * 16);
    acc00 = __builtin_amdgcn_wmma_f32_16x16x32_bf16(false, a0, false, b0,
                                                    (short)0, acc00, false, false);
    acc01 = __builtin_amdgcn_wmma_f32_16x16x32_bf16(false, a0, false, b1,
                                                    (short)0, acc01, false, false);
    acc10 = __builtin_amdgcn_wmma_f32_16x16x32_bf16(false, a1, false, b0,
                                                    (short)0, acc10, false, false);
    acc11 = __builtin_amdgcn_wmma_f32_16x16x32_bf16(false, a1, false, b1,
                                                    (short)0, acc11, false, false);
  }

  // C/D layout: VGPR r -> M = r + 8*hi within 16-tile, N = lane&15
  auto store_tile = [&](const v8f& acc, int mOff, int nOff) {
#pragma unroll
    for (int r = 0; r < 8; ++r) {
      const int mm = mBase + mOff + hi * 8 + r;
      const int nn = nBase + nOff + l15;
      if (mm < M && nn < N) {
        float v = acc[r];
        if (bias) v += bias[nn];
        if (Cin)  v += Cin[(size_t)mm * ldcin + nn];
        Cout[(size_t)mm * ldc + nn] = v;
      }
    }
  };
  store_tile(acc00, 0, 0);
  store_tile(acc01, 0, 16);
  store_tile(acc10, 16, 0);
  store_tile(acc11, 16, 16);
}

// ---------------------------------------------------------------------------
// Elementwise / gather / small kernels
// ---------------------------------------------------------------------------
__global__ void k_cvt(const float* __restrict__ s, bf16_t* __restrict__ d, long n) {
  long i = (long)blockIdx.x * blockDim.x + threadIdx.x;
  if (i < n) d[i] = (bf16_t)s[i];
}

// convert n valid elements, zero-fill up to ntotal (row padding for GEMM)
__global__ void k_cvt_pad(const float* __restrict__ s, bf16_t* __restrict__ d,
                          long n, long ntotal) {
  long i = (long)blockIdx.x * blockDim.x + threadIdx.x;
  if (i < ntotal) d[i] = (i < n) ? (bf16_t)s[i] : (bf16_t)0.0f;
}

__global__ void k_cvt2d(const float* __restrict__ s, long sld,
                        bf16_t* __restrict__ d, long dld, int rows, int cols) {
  long i = (long)blockIdx.x * blockDim.x + threadIdx.x;
  if (i >= (long)rows * cols) return;
  int r = (int)(i / cols), c = (int)(i % cols);
  d[(size_t)r * dld + c] = (bf16_t)s[(size_t)r * sld + c];
}

__global__ void k_gather_enc(const int* __restrict__ x, const float* __restrict__ emb,
                             bf16_t* __restrict__ we, bf16_t* __restrict__ werev) {
  long i = (long)blockIdx.x * blockDim.x + threadIdx.x;
  const long total = (long)B_ * T_ * NI_;
  if (i >= total) return;
  int ii = (int)(i % NI_);
  long bt = i / NI_;
  int t = (int)(bt % T_), b = (int)(bt / T_);
  bf16_t v = (bf16_t)emb[(size_t)x[bt] * NI_ + ii];
  we[i] = v;
  werev[((size_t)b * T_ + (T_ - 1 - t)) * NI_ + ii] = v;
}

__global__ void k_demb(const int* __restrict__ x, const float* __restrict__ emb,
                       const float* __restrict__ suf, bf16_t* __restrict__ out) {
  long i = (long)blockIdx.x * blockDim.x + threadIdx.x;
  const long total = (long)B_ * T_ * NI_;
  if (i >= total) return;
  int ii = (int)(i % NI_);
  long bt = i / NI_;
  int b = (int)(bt / T_);
  out[i] = (bf16_t)(emb[(size_t)x[bt] * NI_ + ii] + suf[(size_t)b * NI_ + ii]);
}

__global__ void k_state_zero(float* __restrict__ c, bf16_t* __restrict__ hbf, int n) {
  int i = blockIdx.x * blockDim.x + threadIdx.x;
  if (i < n) { c[i] = 0.0f; hbf[i] = (bf16_t)0.0f; }
}

__global__ void k_dec_init(const float* __restrict__ root, float* __restrict__ c,
                           bf16_t* __restrict__ hbf, int n) {
  int i = blockIdx.x * blockDim.x + threadIdx.x;
  if (i < n) { float r = root[i]; c[i] = r; hbf[i] = (bf16_t)tanhf(r); }
}

__device__ __forceinline__ float sigf(float x) { return 1.0f / (1.0f + expf(-x)); }

// gate order (torch): i, f, g, o
__global__ void k_lstm_cell(const float* __restrict__ g, float* __restrict__ c,
                            float* hf32, long hld,
                            bf16_t* __restrict__ hbf,
                            bf16_t* hstep, long hstepld,
                            int H, int n) {
  int i = blockIdx.x * blockDim.x + threadIdx.x;
  if (i >= n) return;
  int b = i / H, j = i % H;
  const float* gr = g + (size_t)b * 4 * H;
  float cn = sigf(gr[H + j]) * c[i] + sigf(gr[j]) * tanhf(gr[2 * H + j]);
  float hn = sigf(gr[3 * H + j]) * tanhf(cn);
  c[i] = cn;
  hbf[i] = (bf16_t)hn;
  if (hf32)  hf32[(size_t)b * hld + j] = hn;
  if (hstep) hstep[(size_t)b * hstepld + j] = (bf16_t)hn;
}

// One block (256 thr) per (b, n): argmin_k ||lat - cb[n,k]||^2
__global__ __launch_bounds__(256) void k_vq(const float* __restrict__ fhs,
                                            const float* __restrict__ cb,
                                            float* __restrict__ mind,
                                            float* __restrict__ indsf,
                                            bf16_t* __restrict__ qbf) {
  const int b = blockIdx.x / ND_;
  const int n = blockIdx.x % ND_;
  const int tid = threadIdx.x;
  __shared__ float lat[DC_];
  __shared__ float sd[256];
  __shared__ int   si[256];
  for (int d = tid; d < DC_; d += 256) lat[d] = fhs[(size_t)b * (2 * EH_) + n * DC_ + d];
  __syncthreads();
  float best = 3.4e38f; int bi = 0;
  for (int k = tid; k < KC_; k += 256) {
    const float* cr = cb + ((size_t)n * KC_ + k) * DC_;
    float s = 0.0f;
    for (int d = 0; d < DC_; ++d) { float df = lat[d] - cr[d]; s += df * df; }
    if (s < best) { best = s; bi = k; }
  }
  sd[tid] = best; si[tid] = bi;
  __syncthreads();
  for (int st = 128; st > 0; st >>= 1) {
    if (tid < st) {
      if (sd[tid + st] < sd[tid] ||
          (sd[tid + st] == sd[tid] && si[tid + st] < si[tid])) {
        sd[tid] = sd[tid + st]; si[tid] = si[tid + st];
      }
    }
    __syncthreads();
  }
  if (tid == 0) { mind[blockIdx.x] = sd[0]; indsf[blockIdx.x] = (float)si[0]; }
  __syncthreads();
  const float* cr = cb + ((size_t)n * KC_ + si[0]) * DC_;
  for (int d = tid; d < DC_; d += 256)
    qbf[(size_t)b * (2 * EH_) + n * DC_ + d] = (bf16_t)cr[d];
}

// fwd value: embl == commit == min_dist/D  ->  vq_loss = (1+beta)/D * sum_n min_dist
__global__ void k_vqloss(const float* __restrict__ mind, float* __restrict__ out) {
  int b = threadIdx.x;
  if (b >= B_) return;
  float s = 0.0f;
  for (int n = 0; n < ND_; ++n) s += mind[b * ND_ + n];
  out[b] = s * (1.25f / (float)DC_);
}

__global__ void k_kl(const float* __restrict__ mulv, float* __restrict__ out) {
  int b = threadIdx.x;
  if (b >= B_) return;
  float s = 0.0f;
  for (int j = 0; j < NZ_; ++j) {
    float mu = mulv[(size_t)b * (2 * NZ_) + j];
    float lv = mulv[(size_t)b * (2 * NZ_) + NZ_ + j];
    s += mu * mu + expf(lv) - lv - 1.0f;
  }
  out[b] = 0.5f * s;
}

// ---------------------------------------------------------------------------
extern "C" void kernel_launch(void* const* d_in, const int* in_sizes, int n_in,
                              void* d_out, int out_size, void* d_ws, size_t ws_size,
                              hipStream_t stream) {
  (void)in_sizes; (void)n_in; (void)out_size; (void)ws_size;
  const int*   x        = (const int*)  d_in[0];
  const float* encEmb   = (const float*)d_in[1];
  const float* encWihF  = (const float*)d_in[2];
  const float* encWhhF  = (const float*)d_in[3];
  const float* encBF    = (const float*)d_in[4];
  const float* encWihB  = (const float*)d_in[5];
  const float* encWhhB  = (const float*)d_in[6];
  const float* encBB    = (const float*)d_in[7];
  const float* encLin   = (const float*)d_in[8];
  const float* z2decW   = (const float*)d_in[9];
  const float* z2decB   = (const float*)d_in[10];
  const float* codebook = (const float*)d_in[11];
  const float* suffixW  = (const float*)d_in[12];
  const float* decEmb   = (const float*)d_in[13];
  const float* decWih   = (const float*)d_in[14];
  const float* decWhh   = (const float*)d_in[15];
  const float* decB     = (const float*)d_in[16];
  const float* predW    = (const float*)d_in[17];

  float* out = (float*)d_out;
  const size_t LOGITS_N = (size_t)B_ * T_ * V_;     // 81,920,000
  float* outVq   = out + LOGITS_N;                  // (B,)
  float* outKl   = outVq + B_;                      // (B,)
  float* outInds = outKl + B_;                      // (B,ND) as float

  // bump allocator over d_ws
  char* ws = (char*)d_ws;
  size_t off = 0;
  auto alloc = [&](size_t bytes) -> void* {
    void* p = ws + off;
    off = (off + bytes + 255) & ~(size_t)255;
    return p;
  };

  bf16_t* wEncWihF = (bf16_t*)alloc((size_t)4 * EH_ * NI_ * 2);
  bf16_t* wEncWhhF = (bf16_t*)alloc((size_t)4 * EH_ * EH_ * 2);
  bf16_t* wEncWihB = (bf16_t*)alloc((size_t)4 * EH_ * NI_ * 2);
  bf16_t* wEncWhhB = (bf16_t*)alloc((size_t)4 * EH_ * EH_ * 2);
  bf16_t* wEncLin  = (bf16_t*)alloc((size_t)2 * NZ_ * 2 * EH_ * 2);
  bf16_t* wZ2dec   = (bf16_t*)alloc((size_t)DH_ * NZ_ * 2);
  bf16_t* wSuffix  = (bf16_t*)alloc((size_t)NI_ * 2 * EH_ * 2);
  bf16_t* wDecWih  = (bf16_t*)alloc((size_t)4 * DH_ * NI_ * 2);
  bf16_t* wDecWhh  = (bf16_t*)alloc((size_t)4 * DH_ * DH_ * 2);
  bf16_t* wPred    = (bf16_t*)alloc((size_t)VPAD_ * DH_ * 2);   // zero-padded rows

  bf16_t* weBf     = (bf16_t*)alloc((size_t)B_ * T_ * NI_ * 2);
  bf16_t* weRevBf  = (bf16_t*)alloc((size_t)B_ * T_ * NI_ * 2);
  bf16_t* dembBf   = (bf16_t*)alloc((size_t)B_ * T_ * NI_ * 2);
  float*  xgF      = (float*) alloc((size_t)B_ * T_ * 4 * EH_ * 4);
  float*  xgB      = (float*) alloc((size_t)B_ * T_ * 4 * EH_ * 4);
  float*  xgD      = (float*) alloc((size_t)B_ * T_ * 4 * DH_ * 4);
  float*  gbuf     = (float*) alloc((size_t)B_ * 4 * DH_ * 4);
  float*  cbuf     = (float*) alloc((size_t)B_ * DH_ * 4);
  bf16_t* hbf      = (bf16_t*)alloc((size_t)B_ * DH_ * 2);
  float*  fhs      = (float*) alloc((size_t)B_ * 2 * EH_ * 4);
  bf16_t* fhsBf    = (bf16_t*)alloc((size_t)B_ * 2 * EH_ * 2);
  float*  mulv     = (float*) alloc((size_t)B_ * 2 * NZ_ * 4);
  bf16_t* muBf     = (bf16_t*)alloc((size_t)B_ * NZ_ * 2);
  float*  root     = (float*) alloc((size_t)B_ * DH_ * 4);
  float*  mind     = (float*) alloc((size_t)B_ * ND_ * 4);
  bf16_t* qBf      = (bf16_t*)alloc((size_t)B_ * 2 * EH_ * 2);
  float*  suffix   = (float*) alloc((size_t)B_ * NI_ * 4);
  bf16_t* decOutBf = (bf16_t*)alloc((size_t)B_ * T_ * DH_ * 2);

  auto cvt = [&](const float* s, bf16_t* d, long n) {
    k_cvt<<<(unsigned)((n + 255) / 256), 256, 0, stream>>>(s, d, n);
  };
  auto gemm = [&](const bf16_t* A, long lda, const bf16_t* W, const float* bias,
                  const float* Cin, long ldcin, float* Cout, long ldc,
                  int M, int N, int K) {
    dim3 g((unsigned)((N + 127) / 128), (unsigned)((M + 63) / 64));
    k_gemm_bf16<<<g, 256, 0, stream>>>(A, lda, W, bias, Cin, ldcin, Cout, ldc, M, N, K);
  };

  // 1) weights -> bf16 (pred_W zero-padded to VPAD_ rows)
  cvt(encWihF, wEncWihF, (long)4 * EH_ * NI_);
  cvt(encWhhF, wEncWhhF, (long)4 * EH_ * EH_);
  cvt(encWihB, wEncWihB, (long)4 * EH_ * NI_);
  cvt(encWhhB, wEncWhhB, (long)4 * EH_ * EH_);
  cvt(encLin,  wEncLin,  (long)2 * NZ_ * 2 * EH_);
  cvt(z2decW,  wZ2dec,   (long)DH_ * NZ_);
  cvt(suffixW, wSuffix,  (long)NI_ * 2 * EH_);
  cvt(decWih,  wDecWih,  (long)4 * DH_ * NI_);
  cvt(decWhh,  wDecWhh,  (long)4 * DH_ * DH_);
  {
    long nv = (long)V_ * DH_, nt = (long)VPAD_ * DH_;
    k_cvt_pad<<<(unsigned)((nt + 255) / 256), 256, 0, stream>>>(predW, wPred, nv, nt);
  }

  // 2) embedding gather (fwd + time-reversed) -> bf16
  {
    long n = (long)B_ * T_ * NI_;
    k_gather_enc<<<(unsigned)((n + 255) / 256), 256, 0, stream>>>(x, encEmb, weBf, weRevBf);
  }

  // 3) input-gate precompute for both encoder directions
  gemm(weBf,    NI_, wEncWihF, encBF, nullptr, 0, xgF, 4 * EH_, B_ * T_, 4 * EH_, NI_);
  gemm(weRevBf, NI_, wEncWihB, encBB, nullptr, 0, xgB, 4 * EH_, B_ * T_, 4 * EH_, NI_);

  // 4) encoder LSTMs (fwd writes fhs[:, :EH], bwd writes fhs[:, EH:])
  const int nEH = B_ * EH_;
  for (int dir = 0; dir < 2; ++dir) {
    const float*  xg  = dir ? xgB : xgF;
    const bf16_t* whh = dir ? wEncWhhB : wEncWhhF;
    float* hdst = fhs + (dir ? EH_ : 0);
    k_state_zero<<<(nEH + 255) / 256, 256, 0, stream>>>(cbuf, hbf, nEH);
    for (int t = 0; t < T_; ++t) {
      gemm(hbf, EH_, whh, nullptr, xg + (size_t)t * 4 * EH_, (long)T_ * 4 * EH_,
           gbuf, 4 * EH_, B_, 4 * EH_, EH_);
      k_lstm_cell<<<(nEH + 255) / 256, 256, 0, stream>>>(
          gbuf, cbuf, hdst, (long)2 * EH_, hbf, nullptr, 0, EH_, nEH);
    }
  }

  // 5) mu/logvar, kl, root
  cvt(fhs, fhsBf, (long)B_ * 2 * EH_);
  gemm(fhsBf, 2 * EH_, wEncLin, nullptr, nullptr, 0, mulv, 2 * NZ_, B_, 2 * NZ_, 2 * EH_);
  k_kl<<<1, B_, 0, stream>>>(mulv, outKl);
  k_cvt2d<<<(B_ * NZ_ + 255) / 256, 256, 0, stream>>>(mulv, 2 * NZ_, muBf, NZ_, B_, NZ_);
  gemm(muBf, NZ_, wZ2dec, z2decB, nullptr, 0, root, DH_, B_, DH_, NZ_);

  // 6) VQ: argmin + q gather + losses
  k_vq<<<B_ * ND_, 256, 0, stream>>>(fhs, codebook, mind, outInds, qBf);
  k_vqloss<<<1, B_, 0, stream>>>(mind, outVq);

  // 7) suffix + decoder inputs
  gemm(qBf, 2 * EH_, wSuffix, nullptr, nullptr, 0, suffix, NI_, B_, NI_, 2 * EH_);
  {
    long n = (long)B_ * T_ * NI_;
    k_demb<<<(unsigned)((n + 255) / 256), 256, 0, stream>>>(x, decEmb, suffix, dembBf);
  }
  gemm(dembBf, NI_, wDecWih, decB, nullptr, 0, xgD, 4 * DH_, B_ * T_, 4 * DH_, NI_);

  // 8) decoder LSTM: h0 = tanh(root), c0 = root
  const int nDH = B_ * DH_;
  k_dec_init<<<(nDH + 255) / 256, 256, 0, stream>>>(root, cbuf, hbf, nDH);
  for (int t = 0; t < T_; ++t) {
    gemm(hbf, DH_, wDecWhh, nullptr, xgD + (size_t)t * 4 * DH_, (long)T_ * 4 * DH_,
         gbuf, 4 * DH_, B_, 4 * DH_, DH_);
    k_lstm_cell<<<(nDH + 255) / 256, 256, 0, stream>>>(
        gbuf, cbuf, nullptr, 0, hbf,
        decOutBf + (size_t)t * DH_, (long)T_ * DH_, DH_, nDH);
  }

  // 9) logits = dec_out @ pred_W^T  (pred rows zero-padded; store guarded)
  gemm(decOutBf, DH_, wPred, nullptr, nullptr, 0, out, V_, B_ * T_, V_, DH_);
}
